// MultiHeadAttention_78426102825512
// MI455X (gfx1250) — matrix operand order repaired and checked by reading, not compiled, not run on hardware
//
#include <hip/hip_runtime.h>

typedef __attribute__((ext_vector_type(16))) _Float16 v16h;
typedef __attribute__((ext_vector_type(8)))  _Float16 v8h;
typedef __attribute__((ext_vector_type(4)))  _Float16 v4h;
typedef __attribute__((ext_vector_type(8)))  float    v8f;
typedef __attribute__((ext_vector_type(4)))  int      v4i;

#define SHUF16(lo, hi) __builtin_shufflevector((lo), (hi), 0,1,2,3,4,5,6,7,8,9,10,11,12,13,14,15)

// Problem dims
static constexpr int kB = 2, kS = 2048, kD = 1024, kH = 16, kHD = 64;

// ---------------------------------------------------------------------------
// Async global->LDS 16-byte copy (CDNA5 GLOBAL_LOAD_ASYNC_TO_LDS_B128,
// tracked by ASYNCcnt). Signature probe: param0 is v4i addrspace(1)*
// (clang prints AS1 as "__device__"), non-const. Falls back to a VGPR
// round-trip copy if the builtin is unavailable.
// ---------------------------------------------------------------------------
#if defined(__has_builtin) && __has_builtin(__builtin_amdgcn_global_load_async_to_lds_b128)
#define HAS_ASYNC_LDS 1
#else
#define HAS_ASYNC_LDS 0
#endif

__device__ __forceinline__ void async_ld16(_Float16* lds_dst, const _Float16* gsrc) {
#if HAS_ASYNC_LDS
  __builtin_amdgcn_global_load_async_to_lds_b128(
      (__attribute__((address_space(1))) v4i*)gsrc,
      (__attribute__((address_space(3))) v4i*)lds_dst, 0, 0);
#else
  *(v8h*)lds_dst = *(const v8h*)gsrc;
#endif
}

__device__ __forceinline__ void wait_async0() {
#if HAS_ASYNC_LDS
#if defined(__has_builtin) && __has_builtin(__builtin_amdgcn_s_wait_asynccnt)
  __builtin_amdgcn_s_wait_asynccnt(0);
#else
  asm volatile("s_wait_asynccnt 0x0" ::: "memory");
#endif
#endif
}

// ---------------------------------------------------------------------------
// Kernel 1: Y = relu(X @ W^T + b), X:[4096,1024] f32, W:[1024,1024] f32.
// Output stored f16 in head-split layout [B,H,S,64].
// Block tile 128(M) x 64(N), BK=32, 8 waves (4 M x 2 N), each wave 32x32.
// ---------------------------------------------------------------------------
__global__ __launch_bounds__(256)
void proj_relu_gemm(const float* __restrict__ X, const float* __restrict__ W,
                    const float* __restrict__ bias, _Float16* __restrict__ Y) {
  constexpr int BM = 128, BN = 64, BK = 32, K = 1024, LDA = BK + 8;
  __shared__ __align__(16) _Float16 As[BM][LDA];   // 10.0 KB
  __shared__ __align__(16) _Float16 Bs[BN][LDA];   //  5.0 KB
  const int tid  = threadIdx.x;
  const int lane = tid & 31, wave = tid >> 5;
  const int half = lane >> 4, l16 = lane & 15;
  const int m0 = blockIdx.x * BM, n0 = blockIdx.y * BN;
  const int wm = (wave & 3) * 32, wn = (wave >> 2) * 32;

  v8f acc[2][2] = {};

  for (int k0 = 0; k0 < K; k0 += BK) {
    __syncthreads();
    // Stage A tile (128x32 f32 -> f16), 4 float4 per thread
#pragma unroll
    for (int i = 0; i < 4; ++i) {
      int idx = tid + i * 256;
      int row = idx >> 3, col = (idx & 7) * 4;
      float4 f = *(const float4*)(X + (size_t)(m0 + row) * K + k0 + col);
      *(v4h*)&As[row][col] =
          (v4h){(_Float16)f.x, (_Float16)f.y, (_Float16)f.z, (_Float16)f.w};
    }
    // Stage B tile (64x32 f32 -> f16), 2 float4 per thread
#pragma unroll
    for (int i = 0; i < 2; ++i) {
      int idx = tid + i * 256;
      int row = idx >> 3, col = (idx & 7) * 4;
      float4 f = *(const float4*)(W + (size_t)(n0 + row) * K + k0 + col);
      *(v4h*)&Bs[row][col] =
          (v4h){(_Float16)f.x, (_Float16)f.y, (_Float16)f.z, (_Float16)f.w};
    }
    if (k0 + BK < K) {
      if (tid < 128) __builtin_prefetch(X + (size_t)(m0 + tid) * K + k0 + BK, 0, 3);
      if (tid < 64)  __builtin_prefetch(W + (size_t)(n0 + tid) * K + k0 + BK, 0, 3);
    }
    __syncthreads();

#pragma unroll
    for (int mt = 0; mt < 2; ++mt) {
      // A fragment 16x32 (ISA 7.12.2): lane row = l16, k chunks at half*8 and 16+half*8
      int row = wm + mt * 16 + l16;
      v8h alo = *(const v8h*)&As[row][half * 8];
      v8h ahi = *(const v8h*)&As[row][16 + half * 8];
      v16h af = SHUF16(alo, ahi);
#pragma unroll
      for (int nt = 0; nt < 2; ++nt) {
        // B fragment 32x16: lane col = l16, k chunks at half*8 and 16+half*8
        int coln = wn + nt * 16 + l16;
        v8h blo = *(const v8h*)&Bs[coln][half * 8];
        v8h bhi = *(const v8h*)&Bs[coln][16 + half * 8];
        v16h bf = SHUF16(blo, bhi);
        acc[mt][nt] = __builtin_amdgcn_wmma_f32_16x16x32_f16(
            false, af, false, bf, (short)0, acc[mt][nt], false, false);
      }
    }
  }

  // Epilogue: bias + relu, store f16 head-split [B,H,S,64]
#pragma unroll
  for (int mt = 0; mt < 2; ++mt) {
#pragma unroll
    for (int nt = 0; nt < 2; ++nt) {
      int n = n0 + wn + nt * 16 + l16;
      float bv = bias[n];
      int hh = n >> 6, e = n & 63;
#pragma unroll
      for (int r = 0; r < 8; ++r) {
        int m = m0 + wm + mt * 16 + r + 8 * half;  // C layout: row = r + 8*(lane/16)
        int b = m >> 11, s = m & 2047;
        float v = fmaxf(acc[mt][nt][r] + bv, 0.0f);
        Y[(((size_t)b * kH + hh) * kS + s) * kHD + e] = (_Float16)v;
      }
    }
  }
}

// ---------------------------------------------------------------------------
// Kernel 2: flash attention per (b,h). Block = 128 query rows, 8 waves,
// each wave owns 16 query rows. 64-key tiles, online softmax.
// K tiles staged via async global->LDS; V staged transposed through VGPRs.
// ---------------------------------------------------------------------------
__global__ __launch_bounds__(256)
void flash_attn(const _Float16* __restrict__ Qh, const _Float16* __restrict__ Kh,
                const _Float16* __restrict__ Vh, const int* __restrict__ amask,
                float* __restrict__ Oa) {
  constexpr int LDK = 72;
  __shared__ __align__(16) _Float16 Ks[64][LDK];     // 9 KB   [key][hd]
  __shared__ __align__(16) _Float16 Vt[64][LDK];     // 9 KB   [hd][key] (transposed)
  __shared__ __align__(16) _Float16 Ps[8][16][64];   // 16 KB  per-wave P tiles

  const int tid  = threadIdx.x;
  const int lane = tid & 31, wave = tid >> 5;
  const int half = lane >> 4, l16 = lane & 15;
  const int bh = blockIdx.y;            // b*16 + h
  const int b  = bh >> 4;
  const int qbase = blockIdx.x * 128 + wave * 16;
  const size_t hbase = (size_t)bh * kS * kHD;

  // Q fragments: 16 rows x 64 hd = two 16x32 A fragments, loaded once.
  v16h qa[2];
  {
    const _Float16* qrow = Qh + hbase + (size_t)(qbase + l16) * kHD;
#pragma unroll
    for (int c = 0; c < 2; ++c) {
      v8h lo = *(const v8h*)(qrow + c * 32 + half * 8);
      v8h hi = *(const v8h*)(qrow + c * 32 + half * 8 + 16);
      qa[c] = SHUF16(lo, hi);
    }
  }

  float mrow[8], lrow[8];
  bool rowon[8];
#pragma unroll
  for (int r = 0; r < 8; ++r) {
    mrow[r] = -3.0e38f;
    lrow[r] = 0.0f;
    rowon[r] = amask[b * kS + qbase + r + 8 * half] != 0;
  }
  v8f o[4] = {};

  for (int k0 = 0; k0 < kS; k0 += 64) {
    __syncthreads();
    // Stage K (async -> LDS) and V transposed (64x64 halfs each)
#pragma unroll
    for (int i = 0; i < 2; ++i) {
      int idx = tid + i * 256;
      int row = idx >> 3, col = (idx & 7) * 8;
      async_ld16(&Ks[row][col], Kh + hbase + (size_t)(k0 + row) * kHD + col);
      v8h vv = *(const v8h*)(Vh + hbase + (size_t)(k0 + row) * kHD + col);
#pragma unroll
      for (int j = 0; j < 8; ++j) Vt[col + j][row] = vv[j];
    }
    if (k0 + 64 < kS) {
      __builtin_prefetch(Kh + hbase + (size_t)(k0 + 64 + (tid >> 2)) * kHD + (tid & 3) * 16, 0, 3);
      __builtin_prefetch(Vh + hbase + (size_t)(k0 + 64 + (tid >> 2)) * kHD + (tid & 3) * 16, 0, 3);
    }
    wait_async0();     // our async tile writes complete (ASYNCcnt -> 0)
    __syncthreads();   // everyone's writes visible

    // S = Q * K^T : 4 key-tiles of 16, hd split in two 32-chunks
    v8f sc[4] = {};
#pragma unroll
    for (int nt = 0; nt < 4; ++nt) {
      int key = nt * 16 + l16;
#pragma unroll
      for (int kc = 0; kc < 2; ++kc) {
        v8h lo = *(const v8h*)&Ks[key][kc * 32 + half * 8];
        v8h hi = *(const v8h*)&Ks[key][kc * 32 + half * 8 + 16];
        v16h bf = SHUF16(lo, hi);
        sc[nt] = __builtin_amdgcn_wmma_f32_16x16x32_f16(
            false, qa[kc], false, bf, (short)0, sc[nt], false, false);
      }
    }

    // Online softmax (rows live across the 16 lanes of a half-wave)
#pragma unroll
    for (int r = 0; r < 8; ++r) {
#pragma unroll
      for (int nt = 0; nt < 4; ++nt) {
        float s = sc[nt][r] * 0.125f;  // 1/sqrt(64)
        sc[nt][r] = rowon[r] ? s : -1.0e9f;
      }
      float v = fmaxf(fmaxf(sc[0][r], sc[1][r]), fmaxf(sc[2][r], sc[3][r]));
#pragma unroll
      for (int off = 1; off < 16; off <<= 1) v = fmaxf(v, __shfl_xor(v, off, 16));
      float mn = fmaxf(mrow[r], v);
      float alpha = __expf(mrow[r] - mn);
      mrow[r] = mn;
      float sum = 0.0f;
#pragma unroll
      for (int nt = 0; nt < 4; ++nt) {
        float p = __expf(sc[nt][r] - mn);
        sc[nt][r] = p;
        sum += p;
      }
#pragma unroll
      for (int off = 1; off < 16; off <<= 1) sum += __shfl_xor(sum, off, 16);
      lrow[r] = lrow[r] * alpha + sum;
#pragma unroll
      for (int nt = 0; nt < 4; ++nt) o[nt][r] *= alpha;
      // Spill P tile (C layout) to per-wave LDS for A-fragment re-read
      int prow = r + 8 * half;
#pragma unroll
      for (int nt = 0; nt < 4; ++nt)
        Ps[wave][prow][nt * 16 + l16] = (_Float16)sc[nt][r];
    }
    // Same-wave LDS store->load dependency: drain DScnt (CDNA5 split counter)
    asm volatile("s_wait_dscnt 0x0" ::: "memory");

    // O += P * V : P as A fragments from LDS, Vt as B fragments
#pragma unroll
    for (int kc = 0; kc < 2; ++kc) {
      const _Float16* pr = &Ps[wave][l16][0];
      v8h lo = *(const v8h*)(pr + kc * 32 + half * 8);
      v8h hi = *(const v8h*)(pr + kc * 32 + half * 8 + 16);
      v16h pa = SHUF16(lo, hi);
#pragma unroll
      for (int nt = 0; nt < 4; ++nt) {
        int e = nt * 16 + l16;
        v8h blo = *(const v8h*)&Vt[e][kc * 32 + half * 8];
        v8h bhi = *(const v8h*)&Vt[e][kc * 32 + half * 8 + 16];
        v16h bv = SHUF16(blo, bhi);
        o[nt] = __builtin_amdgcn_wmma_f32_16x16x32_f16(
            false, pa, false, bv, (short)0, o[nt], false, false);
      }
    }
  }

  // Normalize and store f32 [B,H,S,64]
#pragma unroll
  for (int r = 0; r < 8; ++r) {
    float inv = 1.0f / lrow[r];
    int q = qbase + r + 8 * half;
#pragma unroll
    for (int nt = 0; nt < 4; ++nt)
      Oa[hbase + (size_t)q * kHD + nt * 16 + l16] = o[nt][r] * inv;
  }
}

// ---------------------------------------------------------------------------
// Kernel 3: faithful transpose(0,1,3,2).reshape(B,S,D) remap + residual.
// out[b, s, d] = a[b, h, t, e] + queries[b, s, d]
//   flat = s*D + d ; h = flat>>17 ; e = (flat>>11)&63 ; t = flat&2047
// 4 elements per thread: q/out are float4; the 4 A reads stride by 64 floats.
// ---------------------------------------------------------------------------
__global__ __launch_bounds__(256)
void out_remap_add(const float* __restrict__ A, const float* __restrict__ Q,
                   float* __restrict__ Out, size_t total4) {
  size_t idx = (size_t)blockIdx.x * 256 + threadIdx.x;
  if (idx >= total4) return;
  size_t i = idx * 4;
  size_t b = i >> 21;                       // S*D = 2^21
  int flat = (int)(i & ((1u << 21) - 1));
  int hh = flat >> 17;                      // H*hd*S chunk = 64*2048 = 2^17
  int rr = flat & 131071;
  int e = rr >> 11;                         // /2048
  int t = rr & 2047;                        // t..t+3 consecutive (i % 2048 <= 2044)
  const float* ap = A + (((size_t)b * kH + hh) * kS + t) * kHD + e;
  float4 qv = *(const float4*)(Q + i);
  float4 ov;
  ov.x = ap[0 * kHD] + qv.x;
  ov.y = ap[1 * kHD] + qv.y;
  ov.z = ap[2 * kHD] + qv.z;
  ov.w = ap[3 * kHD] + qv.w;
  *(float4*)(Out + i) = ov;
}

// ---------------------------------------------------------------------------
extern "C" void kernel_launch(void* const* d_in, const int* in_sizes, int n_in,
                              void* d_out, int out_size, void* d_ws, size_t ws_size,
                              hipStream_t stream) {
  (void)in_sizes; (void)n_in; (void)ws_size;
  const float* queries = (const float*)d_in[0];
  const float* keys    = (const float*)d_in[1];
  const float* values  = (const float*)d_in[2];
  const int*   amask   = (const int*)d_in[3];
  const float* Wq = (const float*)d_in[4];
  const float* bq = (const float*)d_in[5];
  const float* Wk = (const float*)d_in[6];
  const float* bk = (const float*)d_in[7];
  const float* Wv = (const float*)d_in[8];
  const float* bv = (const float*)d_in[9];

  const size_t elems = (size_t)kB * kS * kD;  // 4,194,304
  _Float16* qh = (_Float16*)d_ws;
  _Float16* kh = qh + elems;
  _Float16* vh = kh + elems;
  float* attn_o = (float*)(vh + elems);

  dim3 gb(kB * kS / 128, kD / 64);  // 32 x 16
  proj_relu_gemm<<<gb, 256, 0, stream>>>(queries, Wq, bq, qh);
  proj_relu_gemm<<<gb, 256, 0, stream>>>(keys,    Wk, bk, kh);
  proj_relu_gemm<<<gb, 256, 0, stream>>>(values,  Wv, bv, vh);

  flash_attn<<<dim3(kS / 128, kB * kH), 256, 0, stream>>>(qh, kh, vh, amask, attn_o);

  size_t total4 = (size_t)out_size / 4;
  out_remap_add<<<(unsigned)((total4 + 255) / 256), 256, 0, stream>>>(
      attn_o, queries, (float*)d_out, total4);
}
